// MicroSpeech_80161269613381
// MI455X (gfx1250) — matrix operand order repaired and checked by reading, not compiled, not run on hardware
//
#include <hip/hip_runtime.h>
#include <hip/hip_bf16.h>
#include <math.h>

// ---------------- types ----------------
typedef _Float16 v16h __attribute__((ext_vector_type(16)));
typedef _Float16 v8h  __attribute__((ext_vector_type(8)));
typedef float    v8f  __attribute__((ext_vector_type(8)));

#define LFRAMES 65536
#define SUB 64           // frames per wave in scan kernels
#define WPB 4            // scan waves per block
#define WU  64           // warm-up frames (|lam|^64 < 1e-27, below fp32 resolution)

// packed-weight sub-offsets (in halves)
#define PW1  0           // 6*4  frags (96x128)
#define PC1  12288       // 2*2  frags (32x64)
#define PMLP 14336       // 2*1  frags (32x32)
#define PW2  15360       // 4*1  frags (64x32)
#define PW3  17408       // 16*3 frags (256x96)
#define PACK_HALVES 41984

// workspace byte offsets
#define OFF_U    0ull                               // (L,96) f32 : BxR1|BxI1|DX
#define OFF_U2   25165824ull                        // (L,64) f32 : BxR2|BxI2
#define OFF_Z    41943040ull                        // (L,32) f16 : post-mlp selu
#define OFF_PACK 46137344ull                        // packed f16 weights
#define OFF_LAM  (OFF_PACK + 2ull*PACK_HALVES)      // lam1R|lam1I|lam2R|lam2I f32
#define WS_NEED  (OFF_LAM + 512ull)

__device__ __forceinline__ v8f wmma16(v16h a, v16h b, v8f c) {
  return __builtin_amdgcn_wmma_f32_16x16x32_f16(false, a, false, b, (short)0, c,
                                                false, false);
}

__device__ __forceinline__ float selu_f(float x) {
  const float sc = 1.0507009873554805f, al = 1.6732632423543772f;
  return x > 0.f ? sc * x : sc * al * (expf(x) - 1.f);
}

// Pack weight matrix W(M,K) into per-lane A-fragment layout for
// v_wmma_f32_16x16x32_f16 (this file's contiguous k order; B uses the same): frag = mt*Kt+kt, entry = lane*16+j,
//   m = mt*16 + lane%16,  k = kt*32 + 16*(lane/16) + j
template <typename F>
__device__ void packA(_Float16* dst, int Mt, int Kt, int tid, int nthr, F f) {
  int total = Mt * Kt * 512;
  for (int e = tid; e < total; e += nthr) {
    int frag = e >> 9, r = e & 511;
    int kt = frag % Kt, mt = frag / Kt;
    int lane = r >> 4, j = r & 15;
    int m = mt * 16 + (lane & 15);
    int k = kt * 32 + 16 * (lane >> 4) + j;
    dst[e] = (_Float16)f(m, k);
  }
}

// ---------------- K0: parameter prep + weight packing ----------------
__global__ void __launch_bounds__(256) k0_prep(
    const float* __restrict__ b1r, const float* __restrict__ b1i,
    const float* __restrict__ nu1, const float* __restrict__ th1,
    const float* __restrict__ c1r, const float* __restrict__ c1i,
    const float* __restrict__ d1,  const float* __restrict__ mlp_w,
    const float* __restrict__ b2r, const float* __restrict__ b2i,
    const float* __restrict__ nu2, const float* __restrict__ th2,
    const float* __restrict__ c2r, const float* __restrict__ c2i,
    const float* __restrict__ d2,
    _Float16* __restrict__ packs, float* __restrict__ lamArr) {
  __shared__ float g1[32], g2[32];
  int tid = threadIdx.x;
  if (tid < 32) {
    float m1 = expf(-expf(nu1[tid])), a1 = expf(th1[tid]);
    lamArr[tid]      = m1 * cosf(a1);
    lamArr[32 + tid] = m1 * sinf(a1);
    g1[tid] = sqrtf(fmaxf(0.f, 1.f - m1 * m1));
    float m2 = expf(-expf(nu2[tid])), a2 = expf(th2[tid]);
    lamArr[64 + tid] = m2 * cosf(a2);
    lamArr[96 + tid] = m2 * sinf(a2);
    g2[tid] = sqrtf(fmaxf(0.f, 1.f - m2 * m2));
  }
  __syncthreads();
  const int NT = blockDim.x;
  // [gamma1*B1r ; gamma1*B1i ; D1]  (96 x 128)
  packA(packs + PW1, 6, 4, tid, NT, [&](int m, int k) -> float {
    if (m < 32) return g1[m] * b1r[m * 128 + k];
    if (m < 64) return g1[m - 32] * b1i[(m - 32) * 128 + k];
    return d1[(m - 64) * 128 + k];
  });
  // [C1r | -C1i]  (32 x 64)
  packA(packs + PC1, 2, 2, tid, NT, [&](int m, int k) -> float {
    return (k < 32) ? c1r[m * 32 + k] : -c1i[m * 32 + (k - 32)];
  });
  // mlp_w^T  (32 x 32)
  packA(packs + PMLP, 2, 1, tid, NT, [&](int m, int k) -> float {
    return mlp_w[k * 32 + m];
  });
  // [gamma2*B2r ; gamma2*B2i]  (64 x 32)
  packA(packs + PW2, 4, 1, tid, NT, [&](int m, int k) -> float {
    return (m < 32) ? g2[m] * b2r[m * 32 + k]
                    : g2[m - 32] * b2i[(m - 32) * 32 + k];
  });
  // [C2r | -C2i | D2]  (256 x 96)
  packA(packs + PW3, 16, 3, tid, NT, [&](int m, int k) -> float {
    if (k < 32) return c2r[m * 32 + k];
    if (k < 64) return -c2i[m * 32 + (k - 32)];
    return d2[m * 32 + (k - 64)];
  });
}

// ---------------- K1: U(L,96) = [gB1r; gB1i; D1](96x128) @ X(128xL) ----------
__global__ void __launch_bounds__(256) k1_gemm(
    const float* __restrict__ X, const _Float16* __restrict__ packW1,
    float* __restrict__ U) {
  int lane = threadIdx.x & 31;
  int wave = threadIdx.x >> 5;
  int tb = (blockIdx.x * 8 + wave) * 16;   // 16 frames per wave
  int t  = tb + (lane & 15);
  int hi = lane >> 4;
  // B fragments from X (frame t is contiguous: order='F'); fp32 -> f16
  v16h b[4];
  const float* src = X + (size_t)t * 128 + hi * 16;
#pragma unroll
  for (int kt = 0; kt < 4; ++kt) {
    const float4* s4 = (const float4*)(src + kt * 32);
    v16h bb;
#pragma unroll
    for (int i = 0; i < 4; ++i) {
      float4 f = s4[i];
      bb[4 * i + 0] = (_Float16)f.x; bb[4 * i + 1] = (_Float16)f.y;
      bb[4 * i + 2] = (_Float16)f.z; bb[4 * i + 3] = (_Float16)f.w;
    }
    b[kt] = bb;
  }
  const v16h* A = (const v16h*)packW1;
#pragma unroll
  for (int mt = 0; mt < 6; ++mt) {
    v8f acc = {};
#pragma unroll
    for (int kt = 0; kt < 4; ++kt)
      acc = wmma16(A[(mt * 4 + kt) * 32 + lane], b[kt], acc);
    // D: lane -> frame t (=N), rows j = mt*16 + 8*hi + r : contiguous
    float* dst = U + (size_t)t * 96 + mt * 16 + hi * 8;
    ((float4*)dst)[0] = make_float4(acc[0], acc[1], acc[2], acc[3]);
    ((float4*)dst)[1] = make_float4(acc[4], acc[5], acc[6], acc[7]);
  }
}

// ---------------- K2: scan1 + C1 + selu + mlp + selu + B2 -------------------
__global__ void __launch_bounds__(32 * WPB) k2_cell1(
    const float* __restrict__ U, const _Float16* __restrict__ packC1,
    const _Float16* __restrict__ packMLP, const _Float16* __restrict__ packW2,
    const float* __restrict__ lamArr, const float* __restrict__ mlp_b,
    float* __restrict__ U2, _Float16* __restrict__ Z) {
  __shared__ _Float16 hsS_[WPB][SUB * 64];   // hs1 (f16); later z1/z2 staging
  __shared__ _Float16 stS_[WPB][SUB * 32];   // GEMM epilogue staging (f16)
  int lane = threadIdx.x & 31;
  int wave = threadIdx.x >> 5;
  int hi = lane >> 4, lo = lane & 15;
  _Float16* hs = hsS_[wave];
  _Float16* st = stS_[wave];
  int t0 = (blockIdx.x * WPB + wave) * SUB;

  // ---- scan (lane = hidden channel), warm-up then main ----
  float lamR = lamArr[lane], lamI = lamArr[32 + lane];
  float hR = 0.f, hI = 0.f;
  int ts = t0 - WU; if (ts < 0) ts = 0;
#pragma unroll 4
  for (int t = ts; t < t0; ++t) {
    const float* u = U + (size_t)t * 96;
    float uR = u[lane], uI = u[32 + lane];
    float nR = fmaf(lamR, hR, fmaf(-lamI, hI, uR));
    float nI = fmaf(lamR, hI, fmaf(lamI, hR, uI));
    hR = nR; hI = nI;
  }
#pragma unroll 4
  for (int tl = 0; tl < SUB; ++tl) {
    const float* u = U + (size_t)(t0 + tl) * 96;
    float uR = u[lane], uI = u[32 + lane];
    float nR = fmaf(lamR, hR, fmaf(-lamI, hI, uR));
    float nI = fmaf(lamR, hI, fmaf(lamI, hR, uI));
    hR = nR; hI = nI;
    hs[tl * 64 + lane]      = (_Float16)hR;
    hs[tl * 64 + 32 + lane] = (_Float16)hI;
  }
  // ---- y1 = [C1r|-C1i](32x64) @ hs^T ----
  const v16h* AC1 = (const v16h*)packC1;
  for (int nt = 0; nt < SUB / 16; ++nt) {
#pragma unroll
    for (int mt = 0; mt < 2; ++mt) {
      v8f acc = {};
#pragma unroll
      for (int kt = 0; kt < 2; ++kt) {
        v16h bfr = *(const v16h*)(hs + (nt * 16 + lo) * 64 + kt * 32 + hi * 16);
        acc = wmma16(AC1[(mt * 2 + kt) * 32 + lane], bfr, acc);
      }
      v8h h8;
#pragma unroll
      for (int r = 0; r < 8; ++r) h8[r] = (_Float16)acc[r];
      *(v8h*)(st + (nt * 16 + lo) * 32 + mt * 16 + hi * 8) = h8;
    }
  }
  // ---- z1 = selu(y1 + DX) -> hs[0 .. SUB*32) ----
  for (int e = lane; e < SUB * 32; e += 32) {
    int fr = e >> 5, ch = e & 31;
    float v = (float)st[e] + U[(size_t)(t0 + fr) * 96 + 64 + ch];
    hs[e] = (_Float16)selu_f(v);
  }
  // ---- z2pre = mlp_w^T @ z1^T ----
  const v16h* AM = (const v16h*)packMLP;
  for (int nt = 0; nt < SUB / 16; ++nt) {
#pragma unroll
    for (int mt = 0; mt < 2; ++mt) {
      v16h bfr = *(const v16h*)(hs + (nt * 16 + lo) * 32 + hi * 16);
      v8f acc = {};
      acc = wmma16(AM[mt * 32 + lane], bfr, acc);
      v8h h8;
#pragma unroll
      for (int r = 0; r < 8; ++r) h8[r] = (_Float16)acc[r];
      *(v8h*)(st + (nt * 16 + lo) * 32 + mt * 16 + hi * 8) = h8;
    }
  }
  // ---- z2 = selu(z2pre + bias) -> hs[SUB*32 ..) and global Z ----
  for (int e = lane; e < SUB * 32; e += 32) {
    int fr = e >> 5, ch = e & 31;
    _Float16 hv = (_Float16)selu_f((float)st[e] + mlp_b[ch]);
    hs[SUB * 32 + e] = hv;
    Z[(size_t)(t0 + fr) * 32 + ch] = hv;
  }
  // ---- U2 = [gB2r; gB2i](64x32) @ z2^T ----
  const v16h* AW2 = (const v16h*)packW2;
  for (int nt = 0; nt < SUB / 16; ++nt) {
#pragma unroll
    for (int mt = 0; mt < 4; ++mt) {
      v16h bfr = *(const v16h*)(hs + SUB * 32 + (nt * 16 + lo) * 32 + hi * 16);
      v8f acc = {};
      acc = wmma16(AW2[mt * 32 + lane], bfr, acc);
      float* dst = U2 + (size_t)(t0 + nt * 16 + lo) * 64 + mt * 16 + hi * 8;
      ((float4*)dst)[0] = make_float4(acc[0], acc[1], acc[2], acc[3]);
      ((float4*)dst)[1] = make_float4(acc[4], acc[5], acc[6], acc[7]);
    }
  }
}

// ---------------- K3: scan2 + [C2r|-C2i|D2](256x96) @ [hsR|hsI|z]^T ----------
__global__ void __launch_bounds__(32 * WPB) k3_cell2(
    const float* __restrict__ U2, const _Float16* __restrict__ Z,
    const _Float16* __restrict__ packW3, const float* __restrict__ lamArr,
    float* __restrict__ out) {
  __shared__ _Float16 hsS_[WPB][SUB * 96];   // rows 0-31 hsR, 32-63 hsI, 64-95 z
  int lane = threadIdx.x & 31;
  int wave = threadIdx.x >> 5;
  int hi = lane >> 4, lo = lane & 15;
  _Float16* hs = hsS_[wave];
  int t0 = (blockIdx.x * WPB + wave) * SUB;

  float lamR = lamArr[64 + lane], lamI = lamArr[96 + lane];
  float hR = 0.f, hI = 0.f;
  int ts = t0 - WU; if (ts < 0) ts = 0;
#pragma unroll 4
  for (int t = ts; t < t0; ++t) {
    const float* u = U2 + (size_t)t * 64;
    float uR = u[lane], uI = u[32 + lane];
    float nR = fmaf(lamR, hR, fmaf(-lamI, hI, uR));
    float nI = fmaf(lamR, hI, fmaf(lamI, hR, uI));
    hR = nR; hI = nI;
  }
#pragma unroll 4
  for (int tl = 0; tl < SUB; ++tl) {
    const float* u = U2 + (size_t)(t0 + tl) * 64;
    float uR = u[lane], uI = u[32 + lane];
    float nR = fmaf(lamR, hR, fmaf(-lamI, hI, uR));
    float nI = fmaf(lamR, hI, fmaf(lamI, hR, uI));
    hR = nR; hI = nI;
    hs[tl * 96 + lane]      = (_Float16)hR;
    hs[tl * 96 + 32 + lane] = (_Float16)hI;
  }
  // copy z (f16) into K-rows 64..95 as dwords
  const unsigned* zs = (const unsigned*)(Z + (size_t)t0 * 32);
  unsigned* hd = (unsigned*)hs;
  for (int i = lane; i < SUB * 16; i += 32) {
    int fr = i >> 4, pr = i & 15;
    hd[fr * 48 + 32 + pr] = zs[i];
  }
  // output GEMM: Mt=16 (256 out channels), Kt=3, Nt=SUB/16
  const v16h* A3 = (const v16h*)packW3;
  for (int mt = 0; mt < 16; ++mt) {
    v16h a0 = A3[(mt * 3 + 0) * 32 + lane];
    v16h a1 = A3[(mt * 3 + 1) * 32 + lane];
    v16h a2 = A3[(mt * 3 + 2) * 32 + lane];
    for (int nt = 0; nt < SUB / 16; ++nt) {
      const _Float16* base = hs + (nt * 16 + lo) * 96;
      v8f acc = {};
      acc = wmma16(a0, *(const v16h*)(base + 0 * 32 + hi * 16), acc);
      acc = wmma16(a1, *(const v16h*)(base + 1 * 32 + hi * 16), acc);
      acc = wmma16(a2, *(const v16h*)(base + 2 * 32 + hi * 16), acc);
      // D: lane -> column t (=N), rows o = mt*16 + 8*hi + r; out is (256, L)
      size_t tcol = (size_t)(t0 + nt * 16 + lo);
      int ob = mt * 16 + hi * 8;
#pragma unroll
      for (int r = 0; r < 8; ++r)
        out[(size_t)(ob + r) * LFRAMES + tcol] = acc[r];
    }
  }
}

// ---------------- host-side launcher ----------------
extern "C" void kernel_launch(void* const* d_in, const int* in_sizes, int n_in,
                              void* d_out, int out_size, void* d_ws,
                              size_t ws_size, hipStream_t stream) {
  (void)in_sizes; (void)n_in; (void)out_size;
  if (ws_size < WS_NEED) return;
  const float* X     = (const float*)d_in[0];
  const float* b1r   = (const float*)d_in[1];
  const float* b1i   = (const float*)d_in[2];
  const float* nu1   = (const float*)d_in[3];
  const float* th1   = (const float*)d_in[4];
  const float* c1r   = (const float*)d_in[5];
  const float* c1i   = (const float*)d_in[6];
  const float* d1    = (const float*)d_in[7];
  const float* mlp_w = (const float*)d_in[8];
  const float* mlp_b = (const float*)d_in[9];
  const float* b2r   = (const float*)d_in[10];
  const float* b2i   = (const float*)d_in[11];
  const float* nu2   = (const float*)d_in[12];
  const float* th2   = (const float*)d_in[13];
  const float* c2r   = (const float*)d_in[14];
  const float* c2i   = (const float*)d_in[15];
  const float* d2    = (const float*)d_in[16];

  char* ws = (char*)d_ws;
  float*     U      = (float*)(ws + OFF_U);
  float*     U2     = (float*)(ws + OFF_U2);
  _Float16*  Z      = (_Float16*)(ws + OFF_Z);
  _Float16*  packs  = (_Float16*)(ws + OFF_PACK);
  float*     lamArr = (float*)(ws + OFF_LAM);

  k0_prep<<<1, 256, 0, stream>>>(b1r, b1i, nu1, th1, c1r, c1i, d1, mlp_w,
                                 b2r, b2i, nu2, th2, c2r, c2i, d2,
                                 packs, lamArr);
  k1_gemm<<<LFRAMES / 128, 256, 0, stream>>>(X, packs + PW1, U);
  k2_cell1<<<LFRAMES / (WPB * SUB), 32 * WPB, 0, stream>>>(
      U, packs + PC1, packs + PMLP, packs + PW2, lamArr, mlp_b, U2, Z);
  k3_cell2<<<LFRAMES / (WPB * SUB), 32 * WPB, 0, stream>>>(
      U2, Z, packs + PW3, lamArr, (float*)d_out);
}
